// GLVGradientMatchingModel_77094662963533
// MI455X (gfx1250) — matrix operand order repaired and checked by reading, not compiled, not run on hardware
//
#include <hip/hip_runtime.h>

// Problem constants (from reference): T=1024, S=64, N=512
#define TDIM 1024
#define SDIM 64
#define NDIM 512
#define MDIM (TDIM * SDIM)      // 65536 rows of the GEMM
#define BM 128                  // block tile M
#define BN 128                  // block tile N
#define BK 32                   // K step (matches WMMA f16 K)
#define LDSH_STRIDE 40          // f16 stride; 40 halfs = 20 dwords/row -> conflict-free b128
#define LOG2PI 1.8378770664093453f

typedef __attribute__((ext_vector_type(16))) _Float16 v16h;
typedef __attribute__((ext_vector_type(8)))  _Float16 v8h;
typedef __attribute__((ext_vector_type(4)))  _Float16 v4h;
typedef __attribute__((ext_vector_type(8)))  float    v8f;

// ---------------------------------------------------------------------------
// Prep 1: beta[m][n] = beta_matrix[m][n]*(m!=n) - exp(log_beta_self[n]),
// split into f16 hi/lo planes for the 3-product split-f16 WMMA GEMM.
// ---------------------------------------------------------------------------
__global__ __launch_bounds__(256)
void glv_prep_beta(const float* __restrict__ beta_matrix,
                   const float* __restrict__ log_beta_self,
                   _Float16* __restrict__ b_hi,
                   _Float16* __restrict__ b_lo) {
  int idx = blockIdx.x * 256 + threadIdx.x;        // 0 .. N*N-1
  int m = idx >> 9;
  int n = idx & (NDIM - 1);
  float v = beta_matrix[idx] * (m != n ? 1.0f : 0.0f) - __expf(log_beta_self[n]);
  _Float16 hi = (_Float16)v;
  _Float16 lo = (_Float16)(v - (float)hi);
  b_hi[idx] = hi;
  b_lo[idx] = lo;
}

// ---------------------------------------------------------------------------
// Prep 2: alpha[m] = exp(log_alpha[m]);  count_t[t] = sum_n (times[t] >= mask_times[n])
// Launch with 4 blocks x 256 threads (gid 0..1023).
// ---------------------------------------------------------------------------
__global__ __launch_bounds__(256)
void glv_prep_misc(const float* __restrict__ times,
                   const float* __restrict__ mask_times,
                   const float* __restrict__ log_alpha,
                   float* __restrict__ alpha,
                   float* __restrict__ count_t) {
  int gid = blockIdx.x * 256 + threadIdx.x;
  if (gid < NDIM) alpha[gid] = __expf(log_alpha[gid]);
  float tv = times[gid];
  float c = 0.0f;
  for (int n = 0; n < NDIM; ++n) c += (tv >= mask_times[n]) ? 1.0f : 0.0f;
  count_t[gid] = c;
}

// ---------------------------------------------------------------------------
// Main fused kernel: res = (alpha + (x*tmask) @ beta^T) * tmask, plus Gaussian
// log-prob vs gradient and count_t-weighted loss partial per workgroup.
// Grid: (NDIM/BN, MDIM/BM) = (4, 512)  [N fastest -> x-tile reuse in L2/WGP$]
// 256 threads = 8 waves (2 M x 4 N); wave tile 64x32 = 4x2 WMMA accumulators;
// split-f16: acc += Ah*Bh + Ah*Bl + Al*Bh  (3 wmma / tile / kstep = 24/kstep).
// X tile is converted to f16 hi/lo ONCE at staging time (into two LDS planes);
// A fragments are then raw ds_load_b128, no per-fragment cvt chains.
// ---------------------------------------------------------------------------
__global__ __launch_bounds__(256)
void glv_main(const float* __restrict__ x,          // abundance [M][N]
              const float* __restrict__ grad,       // gradient  [M][N]
              const float* __restrict__ times,      // [T]
              const float* __restrict__ mask_times, // [N]
              const _Float16* __restrict__ b_hi,    // [N][N] f16 hi
              const _Float16* __restrict__ b_lo,    // [N][N] f16 lo
              const float* __restrict__ alpha,      // [N]
              const float* __restrict__ count_t,    // [T]
              const float* __restrict__ data_var,   // [1]
              float* __restrict__ out,              // [1 + M*N] : loss, res
              double* __restrict__ partials) {
  __shared__ _Float16 ldsH[BM * LDSH_STRIDE];   // 10 KB, hi plane
  __shared__ _Float16 ldsL[BM * LDSH_STRIDE];   // 10 KB, lo plane
  __shared__ double   sred[256];

  const int tid   = threadIdx.x;
  const int lane  = tid & 31;
  const int wave  = tid >> 5;
  const int waveM = wave & 1;          // 0..1 -> 64 rows each
  const int waveN = wave >> 1;         // 0..3 -> 32 cols each
  const int l16   = lane & 15;
  const int lhi   = lane >> 4;         // 0/1 (lane half)
  const int col0  = blockIdx.x * BN;   // N fastest in dispatch order
  const int row0  = blockIdx.y * BM;
  const int waveRow = waveM * 64;
  const int waveCol = waveN * 32;

  // ---- loop-invariant staging state (4 float4 chunks per thread) ----------
  int srow[4], skc[4];
  const float* xrp[4];
  float tv4[4];
  _Float16 *stH[4], *stL[4];
#pragma unroll
  for (int q = 0; q < 4; ++q) {
    int c = tid + 256 * q;             // chunk 0..1023
    srow[q] = c >> 3;                  // row in tile 0..127
    skc[q]  = (c & 7) * 4;             // k offset in tile {0,4,...,28}
    int grow = row0 + srow[q];
    xrp[q] = x + (size_t)grow * NDIM + skc[q];
    tv4[q] = times[grow >> 6];         // t = row / S  (constant over k)
    stH[q] = ldsH + srow[q] * LDSH_STRIDE + skc[q];
    stL[q] = ldsL + srow[q] * LDSH_STRIDE + skc[q];
  }

  // ---- loop-invariant fragment pointers -----------------------------------
  const _Float16 *apH[4], *apL[4];
#pragma unroll
  for (int mt = 0; mt < 4; ++mt) {
    int row = waveRow + mt * 16 + l16;
    apH[mt] = ldsH + row * LDSH_STRIDE + lhi * 8;
    apL[mt] = ldsL + row * LDSH_STRIDE + lhi * 8;
  }
  const _Float16 *bpH[2], *bpL[2];
#pragma unroll
  for (int nt = 0; nt < 2; ++nt) {
    int colg = col0 + waveCol + nt * 16 + l16;
    size_t boff = (size_t)colg * NDIM + (size_t)(lhi * 8);
    bpH[nt] = b_hi + boff;
    bpL[nt] = b_lo + boff;
  }

  const v8f vzero = {0.f, 0.f, 0.f, 0.f, 0.f, 0.f, 0.f, 0.f};
  v8f acc[4][2];
#pragma unroll
  for (int mt = 0; mt < 4; ++mt)
#pragma unroll
    for (int nt = 0; nt < 2; ++nt) acc[mt][nt] = vzero;

  // software double-buffer: preload K-step 0 into registers
  float4 xr[4];
#pragma unroll
  for (int q = 0; q < 4; ++q) xr[q] = *(const float4*)(xrp[q]);

  for (int k0 = 0; k0 < NDIM; k0 += BK) {
    __syncthreads();  // previous iteration's LDS reads complete
    // ---- convert staged registers -> masked f16 hi/lo LDS planes ---------
#pragma unroll
    for (int q = 0; q < 4; ++q) {
      const float4 mv = *(const float4*)(mask_times + k0 + skc[q]);
      float m0 = (tv4[q] >= mv.x) ? xr[q].x : 0.0f;
      float m1 = (tv4[q] >= mv.y) ? xr[q].y : 0.0f;
      float m2 = (tv4[q] >= mv.z) ? xr[q].z : 0.0f;
      float m3 = (tv4[q] >= mv.w) ? xr[q].w : 0.0f;
      v4h hv, lv;
      hv[0] = (_Float16)m0; lv[0] = (_Float16)(m0 - (float)hv[0]);
      hv[1] = (_Float16)m1; lv[1] = (_Float16)(m1 - (float)hv[1]);
      hv[2] = (_Float16)m2; lv[2] = (_Float16)(m2 - (float)hv[2]);
      hv[3] = (_Float16)m3; lv[3] = (_Float16)(m3 - (float)hv[3]);
      *(v4h*)stH[q] = hv;
      *(v4h*)stL[q] = lv;
    }
    __syncthreads();

    // ---- kick off next K-step's global loads (overlap with WMMA) ---------
    if (k0 + BK < NDIM) {
#pragma unroll
      for (int q = 0; q < 4; ++q) xr[q] = *(const float4*)(xrp[q] + k0 + BK);
    }

    // ---- B fragments from global (beta is L2-resident) -------------------
    // B 32x16 f16 layout: lane l -> col = l&15, kb = (l>>4)*8, j<8 -> K = kb+j ; j>=8 -> K = kb+16+(j-8).
    v16h bhF[2], blF[2];
#pragma unroll
    for (int nt = 0; nt < 2; ++nt) {
      v8h h0 = *(const v8h*)(bpH[nt] + k0);
      v8h h1 = *(const v8h*)(bpH[nt] + k0 + 16);
      bhF[nt] = __builtin_shufflevector(h0, h1, 0,1,2,3,4,5,6,7,8,9,10,11,12,13,14,15);
      v8h l0 = *(const v8h*)(bpL[nt] + k0);
      v8h l1 = *(const v8h*)(bpL[nt] + k0 + 16);
      blF[nt] = __builtin_shufflevector(l0, l1, 0,1,2,3,4,5,6,7,8,9,10,11,12,13,14,15);
      if (k0 + BK < NDIM) {   // CDNA5 global_prefetch_b8 of next beta chunk
        __builtin_prefetch(bpH[nt] + k0 + BK, 0, 3);
        __builtin_prefetch(bpL[nt] + k0 + BK, 0, 3);
      }
    }

    // ---- A fragments from LDS (pure b128 loads) + 3-product WMMA ---------
    // A 16x32 f16 layout: lane l -> row = l&15, kb = (l>>4)*8,
    //   element j<8 -> K = kb+j ; element j>=8 -> K = kb+16+(j-8).
#pragma unroll
    for (int mt = 0; mt < 4; ++mt) {
      v8h ah0 = *(const v8h*)(apH[mt]);
      v8h ah1 = *(const v8h*)(apH[mt] + 16);
      v16h ah = __builtin_shufflevector(ah0, ah1, 0,1,2,3,4,5,6,7,8,9,10,11,12,13,14,15);
      v8h al0 = *(const v8h*)(apL[mt]);
      v8h al1 = *(const v8h*)(apL[mt] + 16);
      v16h al = __builtin_shufflevector(al0, al1, 0,1,2,3,4,5,6,7,8,9,10,11,12,13,14,15);
#pragma unroll
      for (int nt = 0; nt < 2; ++nt) {
        acc[mt][nt] = __builtin_amdgcn_wmma_f32_16x16x32_f16(
            false, ah, false, bhF[nt], (short)0, acc[mt][nt], false, false);
        acc[mt][nt] = __builtin_amdgcn_wmma_f32_16x16x32_f16(
            false, ah, false, blF[nt], (short)0, acc[mt][nt], false, false);
        acc[mt][nt] = __builtin_amdgcn_wmma_f32_16x16x32_f16(
            false, al, false, bhF[nt], (short)0, acc[mt][nt], false, false);
      }
    }
  }

  // ---- fused epilogue: alpha add, output mask, res store, log-prob, loss --
  const float var = data_var[0];
  const float cA = -0.5f / var;
  const float cB = -0.5f * (__logf(var) + LOG2PI);
  double wsum = 0.0;
#pragma unroll
  for (int mt = 0; mt < 4; ++mt) {
#pragma unroll
    for (int nt = 0; nt < 2; ++nt) {
      int colg = col0 + waveCol + nt * 16 + l16;
      float a    = alpha[colg];
      float mcol = mask_times[colg];
      int rbase = row0 + waveRow + mt * 16 + lhi * 8;  // C: lanes16-31 hold M=i+8
#pragma unroll
      for (int i = 0; i < 8; ++i) {
        int rg = rbase + i;
        int t  = rg >> 6;
        float tv = times[t];
        float r  = (tv >= mcol) ? (acc[mt][nt][i] + a) : 0.0f;
        size_t oi = (size_t)rg * NDIM + colg;
        out[1 + oi] = r;
        float d  = r - grad[oi];
        float lp = cA * d * d + cB;
        wsum += (double)(count_t[t] * lp);
      }
    }
  }
  sred[tid] = wsum;
  __syncthreads();
  for (int s = 128; s > 0; s >>= 1) {
    if (tid < s) sred[tid] += sred[tid + s];
    __syncthreads();
  }
  if (tid == 0) partials[blockIdx.y * gridDim.x + blockIdx.x] = sred[0];
}

// ---------------------------------------------------------------------------
// Final deterministic reduction: loss = -(sum of 2048 block partials)
// ---------------------------------------------------------------------------
__global__ __launch_bounds__(256)
void glv_reduce(const double* __restrict__ partials, float* __restrict__ out) {
  __shared__ double sred[256];
  double s = 0.0;
  for (int i = threadIdx.x; i < 2048; i += 256) s += partials[i];
  sred[threadIdx.x] = s;
  __syncthreads();
  for (int k = 128; k > 0; k >>= 1) {
    if (threadIdx.x < k) sred[threadIdx.x] += sred[threadIdx.x + k];
    __syncthreads();
  }
  if (threadIdx.x == 0) out[0] = (float)(-sred[0]);
}

// ---------------------------------------------------------------------------
extern "C" void kernel_launch(void* const* d_in, const int* in_sizes, int n_in,
                              void* d_out, int out_size, void* d_ws, size_t ws_size,
                              hipStream_t stream) {
  (void)in_sizes; (void)n_in; (void)out_size; (void)ws_size;
  const float* times         = (const float*)d_in[0];
  const float* gradient      = (const float*)d_in[1];
  const float* abundance     = (const float*)d_in[2];
  const float* log_alpha     = (const float*)d_in[3];
  const float* log_beta_self = (const float*)d_in[4];
  const float* beta_matrix   = (const float*)d_in[5];
  const float* data_var      = (const float*)d_in[6];
  const float* mask_times    = (const float*)d_in[7];
  float* out = (float*)d_out;

  // Workspace layout
  char* ws = (char*)d_ws;
  _Float16* b_hi   = (_Float16*)(ws);                 // 512 KB
  _Float16* b_lo   = (_Float16*)(ws + 524288);        // 512 KB
  float*    alpha  = (float*)(ws + 1048576);          // 2 KB
  float*    cnt_t  = (float*)(ws + 1050624);          // 4 KB
  double*   parts  = (double*)(ws + 1054720);         // 16 KB

  glv_prep_beta<<<(NDIM * NDIM) / 256, 256, 0, stream>>>(beta_matrix, log_beta_self, b_hi, b_lo);
  glv_prep_misc<<<TDIM / 256, 256, 0, stream>>>(times, mask_times, log_alpha, alpha, cnt_t);

  dim3 grid(NDIM / BN, MDIM / BM);  // (4, 512): N fastest -> x-tile L2 reuse
  glv_main<<<grid, 256, 0, stream>>>(abundance, gradient, times, mask_times,
                                     b_hi, b_lo, alpha, cnt_t, data_var, out, parts);
  glv_reduce<<<1, 256, 0, stream>>>(parts, out);
}
